// RNNWithMixedProjection_54425825575656
// MI455X (gfx1250) — matrix operand order repaired and checked by reading, not compiled, not run on hardware
//
#include <hip/hip_runtime.h>
#include <hip/hip_bf16.h>
#include <cstdint>
#include <cstddef>

// ---------------------------------------------------------------------------
// Types for CDNA5 WMMA (wave32, 16x16x32 bf16 -> f32)
// ---------------------------------------------------------------------------
typedef __bf16 bf16;
typedef __bf16 bf16x2  __attribute__((ext_vector_type(2)));
typedef __bf16 bf16x4  __attribute__((ext_vector_type(4)));
typedef __bf16 bf16x8  __attribute__((ext_vector_type(8)));
typedef __bf16 bf16x16 __attribute__((ext_vector_type(16)));
typedef float  f32x8   __attribute__((ext_vector_type(8)));

#define B_     64
#define E_     512
#define M_     64
#define V_     32000
#define NSTEP_ 64
#define NROWS  (B_ * NSTEP_)   // 4096 rows into vocab GEMM
#define JSPLIT 8               // V-dimension split for the vocab GEMM

// Scheduler pipeline hints (soft directives; no-ops for correctness).
// mask: 0x008 = MFMA/WMMA, 0x020 = VMEM read, 0x100 = DS read, 0x002 = VALU
#if __has_builtin(__builtin_amdgcn_sched_group_barrier)
#define SCHED_GROUP(mask, cnt) __builtin_amdgcn_sched_group_barrier((mask), (cnt), 0)
#else
#define SCHED_GROUP(mask, cnt) ((void)0)
#endif

__device__ __forceinline__ f32x8 wmma_bf16(bf16x16 a, bf16x16 b, f32x8 c) {
  // v_wmma_f32_16x16x32_bf16
  return __builtin_amdgcn_wmma_f32_16x16x32_bf16(
      /*neg_a=*/false, a, /*neg_b=*/false, b,
      /*c_mod=*/(short)0, c, /*reuse_a=*/false, /*reuse_b=*/false);
}

// A fragment (16x32, bf16) from an LDS tile stored row-major [16 x 512].
// lane&15 = M row, lane>>4 selects the K-half (contiguous order):
//   halves 0..15 -> K = k0 + 16*hi + 0..15
__device__ __forceinline__ bf16x16 lds_a_frag(const bf16* tile, int cl, int hi, int k0) {
  const bf16* p = tile + cl * E_ + k0 + hi * 16;
  bf16x8 lo = *(const bf16x8*)p;
  bf16x8 hh = *(const bf16x8*)(p + 8);
  bf16x16 r;
#pragma unroll
  for (int i = 0; i < 8; ++i) { r[i] = lo[i]; r[i + 8] = hh[i]; }
  return r;
}

// branchless online logsumexp update
__device__ __forceinline__ void lse_update(float& m, float& s, float v) {
  const float nm = fmaxf(m, v);
  s = s * __expf(m - nm) + __expf(v - nm);
  m = nm;
}
__device__ __forceinline__ void lse_merge(float& m, float& s, float om, float os) {
  const float nm = fmaxf(m, om);
  s = s * __expf(m - nm) + os * __expf(om - nm);
  m = nm;
}

// ---------------------------------------------------------------------------
// Kernel 1: gather latent rows, std-norm (ddof=1, no centering), add noise.
// ---------------------------------------------------------------------------
__global__ void prep_kernel(const int* __restrict__ zi,
                            const float* __restrict__ latent,
                            const float* __restrict__ noise,
                            bf16* __restrict__ z0bf,
                            float* __restrict__ zinit) {
  __shared__ float red[8], redq[8], bc[1];
  const int b = blockIdx.x, tid = threadIdx.x;
  const int wid = tid >> 5, lane = tid & 31;
  const float* lat = latent + (size_t)zi[b] * (2 * E_);
  for (int h = 0; h < 2; ++h) {
    float x0 = lat[h * E_ + tid * 2];
    float x1 = lat[h * E_ + tid * 2 + 1];
    float s = x0 + x1, q = x0 * x0 + x1 * x1;
#pragma unroll
    for (int off = 16; off; off >>= 1) { s += __shfl_xor(s, off); q += __shfl_xor(q, off); }
    if (lane == 0) { red[wid] = s; redq[wid] = q; }
    __syncthreads();
    if (tid == 0) {
      float S = 0.f, Q = 0.f;
      for (int w = 0; w < 8; ++w) { S += red[w]; Q += redq[w]; }
      float mean = S * (1.0f / 512.0f);
      float var  = (Q - 512.0f * mean * mean) * (1.0f / 511.0f);
      bc[0] = 0.113f / (1e-5f + sqrtf(var));
    }
    __syncthreads();
    const float sc = bc[0];
    const float n0 = noise[(size_t)b * (2 * E_) + h * E_ + tid * 2];
    const float n1 = noise[(size_t)b * (2 * E_) + h * E_ + tid * 2 + 1];
    const float v0 = x0 * sc + (n0 - 0.5f) * 0.05f;
    const float v1 = x1 * sc + (n1 - 0.5f) * 0.05f;
    if (h == 0) {
      z0bf[b * E_ + tid * 2]     = (bf16)v0;
      z0bf[b * E_ + tid * 2 + 1] = (bf16)v1;
    } else {
      zinit[b * E_ + tid * 2]     = v0;
      zinit[b * E_ + tid * 2 + 1] = v1;
    }
    __syncthreads();
  }
}

// ---------------------------------------------------------------------------
// Kernel 2: fp32 -> bf16 conversion (vocab weights), vectorized x4.
// ---------------------------------------------------------------------------
__global__ void cvt_bf16_kernel(const float* __restrict__ src,
                                bf16* __restrict__ dst, long n) {
  long i4 = ((long)blockIdx.x * blockDim.x + threadIdx.x) * 4;
  if (i4 < n) {
    float4 v = *(const float4*)(src + i4);
    bf16x4 o;
    o[0] = (bf16)v.x; o[1] = (bf16)v.y; o[2] = (bf16)v.z; o[3] = (bf16)v.w;
    *(bf16x4*)(dst + i4) = o;
  }
}

// ---------------------------------------------------------------------------
// Kernel 3: projections  C(64 x 32768) = z0(64x512) . W^T + bias, x4 matrices.
// Per-K-step pipeline hint: 4 VMEM reads -> 8 cvt VALU -> 8 DS reads -> 4 WMMA.
// ---------------------------------------------------------------------------
__global__ void proj_kernel(const bf16* __restrict__ z0bf,
                            const float* __restrict__ w0, const float* __restrict__ w1,
                            const float* __restrict__ w2, const float* __restrict__ w3,
                            const float* __restrict__ b0, const float* __restrict__ b1,
                            const float* __restrict__ b2, const float* __restrict__ b3,
                            bf16* __restrict__ projOut) {
  __shared__ bf16 Alds[B_ * E_];  // 64 rows x 512 K, bf16 = 64KB
  const int tid = threadIdx.x, wid = tid >> 5, lane = tid & 31;
  const int hi = lane >> 4, cl = lane & 15;
  const int mat = blockIdx.y;
  const float* W    = (mat == 0) ? w0 : (mat == 1) ? w1 : (mat == 2) ? w2 : w3;
  const float* bias = (mat == 0) ? b0 : (mat == 1) ? b1 : (mat == 2) ? b2 : b3;
  bf16* out = projOut + (size_t)mat * B_ * (M_ * E_);

  for (int i = tid; i < (B_ * E_) / 8; i += 256)
    ((bf16x8*)Alds)[i] = ((const bf16x8*)z0bf)[i];
  __syncthreads();

  const int colBase = blockIdx.x * 512;
  for (int ch = 0; ch < 4; ++ch) {                 // 4 x 128 columns
    const int c = colBase + ch * 128 + wid * 16 + cl;
    const float* wrow = W + (size_t)c * E_;
    f32x8 acc[4];
#pragma unroll
    for (int t = 0; t < 4; ++t) acc[t] = (f32x8)(0.0f);
#pragma unroll 4
    for (int ks = 0; ks < 16; ++ks) {              // K = 512 in steps of 32
      const int k0 = ks * 32;
      const float* wp = wrow + k0 + hi * 16;
      bf16x16 bfrag;
#pragma unroll
      for (int i = 0; i < 16; ++i) bfrag[i] = (bf16)wp[i];
      bf16x16 af[4];
#pragma unroll
      for (int t = 0; t < 4; ++t) af[t] = lds_a_frag(Alds + t * 16 * E_, cl, hi, k0);
#pragma unroll
      for (int t = 0; t < 4; ++t) acc[t] = wmma_bf16(af[t], bfrag, acc[t]);
      SCHED_GROUP(0x020, 4);   // 4 global b128 reads (fp32 W)
      SCHED_GROUP(0x002, 8);   // 8 v_cvt_pk_bf16_f32
      SCHED_GROUP(0x100, 8);   // 8 ds_load_b128 (all A frags)
      SCHED_GROUP(0x008, 4);   // 4 v_wmma back-to-back
    }
    const float bb = bias[c];
#pragma unroll
    for (int t = 0; t < 4; ++t)
#pragma unroll
      for (int r = 0; r < 8; ++r) {
        const int row = t * 16 + hi * 8 + r;       // batch index
        out[(size_t)row * (M_ * E_) + c] = (bf16)(acc[t][r] + bb);
      }
  }
}

// ---------------------------------------------------------------------------
// Kernel 4: recurrent scan. One WG per batch; ek/ev/tk/tv (256KB bf16) held
// resident in LDS across all 64 steps (320KB WGP LDS makes this possible).
// ---------------------------------------------------------------------------
__global__ void scan_kernel(const float* __restrict__ zinit,
                            const bf16* __restrict__ proj,
                            bf16* __restrict__ zsbf) {
  __shared__ bf16 mats[4 * M_ * E_];     // 256 KB
  __shared__ float zarr[E_], znarr[E_];
  __shared__ float scoreE[M_], scoreT[M_];
  __shared__ float red[8], redq[8], bc[1];
  const int b = blockIdx.x, tid = threadIdx.x;
  const int wid = tid >> 5, lane = tid & 31;

  for (int m = 0; m < 4; ++m) {
    const bf16* src = proj + ((size_t)m * B_ + b) * (M_ * E_);
    bf16* dst = mats + m * (M_ * E_);
    for (int i = tid; i < (M_ * E_) / 8; i += 256)
      ((bf16x8*)dst)[i] = ((const bf16x8*)src)[i];
  }
  zarr[tid * 2]     = zinit[b * E_ + tid * 2];
  zarr[tid * 2 + 1] = zinit[b * E_ + tid * 2 + 1];
  __syncthreads();

  const bf16* ek = mats;
  const bf16* ev = mats + 1 * M_ * E_;
  const bf16* tk = mats + 2 * M_ * E_;
  const bf16* tv = mats + 3 * M_ * E_;

  for (int step = 0; step < NSTEP_; ++step) {
    // ---- std-norm of z ----
    float x0 = zarr[tid * 2], x1 = zarr[tid * 2 + 1];
    float s = x0 + x1, q = x0 * x0 + x1 * x1;
#pragma unroll
    for (int off = 16; off; off >>= 1) { s += __shfl_xor(s, off); q += __shfl_xor(q, off); }
    if (lane == 0) { red[wid] = s; redq[wid] = q; }
    __syncthreads();
    if (tid == 0) {
      float S = 0.f, Q = 0.f;
      for (int w = 0; w < 8; ++w) { S += red[w]; Q += redq[w]; }
      float mean = S * (1.0f / 512.0f);
      float var  = (Q - 512.0f * mean * mean) * (1.0f / 511.0f);
      bc[0] = 0.113f / (1e-5f + sqrtf(var));
    }
    __syncthreads();
    const float sc = bc[0];
    znarr[tid * 2] = x0 * sc; znarr[tid * 2 + 1] = x1 * sc;
    __syncthreads();

    // ---- attention scores: wave w owns m in [8w, 8w+8) ----
    float znreg[16];
#pragma unroll
    for (int i = 0; i < 16; ++i) znreg[i] = znarr[lane * 16 + i];
    for (int j = 0; j < 8; ++j) {
      const int m = wid * 8 + j;
      const bf16* ekp = ek + m * E_ + lane * 16;
      const bf16* tkp = tk + m * E_ + lane * 16;
      bf16x8 e0 = *(const bf16x8*)ekp, e1 = *(const bf16x8*)(ekp + 8);
      bf16x8 t0 = *(const bf16x8*)tkp, t1 = *(const bf16x8*)(tkp + 8);
      float se = 0.f, st = 0.f;
#pragma unroll
      for (int i = 0; i < 8; ++i) {
        se += znreg[i] * (float)e0[i] + znreg[i + 8] * (float)e1[i];
        st += znreg[i] * (float)t0[i] + znreg[i + 8] * (float)t1[i];
      }
#pragma unroll
      for (int off = 16; off; off >>= 1) { se += __shfl_xor(se, off); st += __shfl_xor(st, off); }
      if (lane == 0) { scoreE[m] = se; scoreT[m] = st; }
    }
    __syncthreads();

    // ---- softmax params (redundant per-thread over 64 values) ----
    float mE = -1e30f, mT = -1e30f;
    for (int m = 0; m < M_; ++m) { mE = fmaxf(mE, scoreE[m]); mT = fmaxf(mT, scoreT[m]); }
    float sE = 0.f, sT = 0.f;
    for (int m = 0; m < M_; ++m) { sE += __expf(scoreE[m] - mE); sT += __expf(scoreT[m] - mT); }
    const float rE = 1.0f / sE, rT = 1.0f / sT;

    // ---- weighted sums: thread owns e pair (2t, 2t+1) ----
    const int e0i = tid * 2;
    float em0 = 0.f, em1 = 0.f, zn0 = 0.f, zn1 = 0.f;
    for (int m = 0; m < M_; ++m) {
      const float wE = __expf(scoreE[m] - mE);
      const float wT = __expf(scoreT[m] - mT);
      bf16x2 v = *(const bf16x2*)(ev + m * E_ + e0i);
      bf16x2 u = *(const bf16x2*)(tv + m * E_ + e0i);
      em0 += wE * (float)v.x; em1 += wE * (float)v.y;
      zn0 += wT * (float)u.x; zn1 += wT * (float)u.y;
    }
    em0 *= rE; em1 *= rE; zn0 *= rT; zn1 *= rT;
    const size_t zo = ((size_t)b * NSTEP_ + step) * E_ + e0i;
    zsbf[zo] = (bf16)em0; zsbf[zo + 1] = (bf16)em1;
    __syncthreads();                 // everyone done reading scores/ev/tv
    zarr[e0i] = zn0; zarr[e0i + 1] = zn1;
    __syncthreads();                 // z fully committed for next step
  }
}

// ---------------------------------------------------------------------------
// Kernel 5: vocab GEMM + partial online log-sum-exp.
// grid = (64 row-groups, JSPLIT). Per-K-step pipeline hint:
//   2 VMEM reads (next B frag) -> 8 DS reads (4 A frags) -> 4 WMMAs.
// ---------------------------------------------------------------------------
__global__ void vocab_kernel(const bf16* __restrict__ zsbf,
                             const bf16* __restrict__ vocabbf,
                             const float* __restrict__ vocab_b,
                             float2* __restrict__ lsePart) {
  __shared__ bf16 Alds[64 * E_];             // 64KB: this WG's 64 zs rows
  __shared__ float lseM[64 * 8], lseS[64 * 8];
  const int tid = threadIdx.x, wid = tid >> 5, lane = tid & 31;
  const int hi = lane >> 4, cl = lane & 15;
  const int jv = blockIdx.y;
  const size_t rowBase = (size_t)blockIdx.x * 64;

  const bf16* src = zsbf + rowBase * E_;
  for (int i = tid; i < (64 * E_) / 8; i += 256)
    ((bf16x8*)Alds)[i] = ((const bf16x8*)src)[i];
  __syncthreads();

  float m_[32], s_[32];
#pragma unroll
  for (int i = 0; i < 32; ++i) { m_[i] = -1e30f; s_[i] = 0.0f; }

  const int NITER = V_ / 128;                // 250, exact
  for (int it = jv; it < NITER; it += JSPLIT) {
    const int c = it * 128 + wid * 16 + cl;
    const bf16* bp = vocabbf + (size_t)c * E_;
    if (it + JSPLIT < NITER)                 // hint this lane's next chunk
      __builtin_prefetch(bp + (size_t)128 * JSPLIT * E_, 0, 0);
    f32x8 acc[4];
#pragma unroll
    for (int t = 0; t < 4; ++t) acc[t] = (f32x8)(0.0f);
    bf16x16 bfrag = *(const bf16x16*)(bp + hi * 16);
#pragma unroll
    for (int ks = 0; ks < 16; ++ks) {
      const int k0 = ks * 32;
      bf16x16 bnext = bfrag;
      if (ks < 15) bnext = *(const bf16x16*)(bp + (ks + 1) * 32 + hi * 16);
      bf16x16 af[4];
#pragma unroll
      for (int t = 0; t < 4; ++t) af[t] = lds_a_frag(Alds + t * 16 * E_, cl, hi, k0);
#pragma unroll
      for (int t = 0; t < 4; ++t) acc[t] = wmma_bf16(af[t], bfrag, acc[t]);
      bfrag = bnext;
      SCHED_GROUP(0x020, 2);   // 2 global b128 reads (next B frag)
      SCHED_GROUP(0x100, 8);   // 8 ds_load_b128 (all 4 A frags)
      SCHED_GROUP(0x008, 4);   // 4 v_wmma back-to-back
    }
    const float bias = vocab_b[c];
#pragma unroll
    for (int t = 0; t < 4; ++t)
#pragma unroll
      for (int r = 0; r < 8; ++r)
        lse_update(m_[t * 8 + r], s_[t * 8 + r], acc[t][r] + bias);
  }

  // combine the 16 lanes (same hi) that share each row's column subset
#pragma unroll
  for (int off = 1; off < 16; off <<= 1) {
#pragma unroll
    for (int slot = 0; slot < 32; ++slot) {
      const float om = __shfl_xor(m_[slot], off);
      const float os = __shfl_xor(s_[slot], off);
      lse_merge(m_[slot], s_[slot], om, os);
    }
  }
  if (cl == 0) {
#pragma unroll
    for (int slot = 0; slot < 32; ++slot) {
      const int t = slot >> 3, r = slot & 7;
      const int rw = t * 16 + hi * 8 + r;          // row within WG
      lseM[rw * 8 + wid] = m_[slot];
      lseS[rw * 8 + wid] = s_[slot];
    }
  }
  __syncthreads();

  // per-row combine across 8 waves, write the partial for this V-chunk
  if (tid < 64) {
    float M = -1e30f, S = 0.0f;
    for (int w = 0; w < 8; ++w)
      lse_merge(M, S, lseM[tid * 8 + w], lseS[tid * 8 + w]);
    const size_t row = rowBase + tid;
    lsePart[row * JSPLIT + jv] = make_float2(M, S);
  }
}

// ---------------------------------------------------------------------------
// Kernel 6: combine JSPLIT partials per row, target-token logit, write yp.
// ---------------------------------------------------------------------------
__global__ void finish_kernel(const bf16* __restrict__ zsbf,
                              const bf16* __restrict__ vocabbf,
                              const float* __restrict__ vocab_b,
                              const int* __restrict__ y,
                              const float2* __restrict__ lsePart,
                              float* __restrict__ out) {
  const int row = blockIdx.x * blockDim.x + threadIdx.x;
  if (row >= NROWS) return;
  float M = -1e30f, S = 0.0f;
#pragma unroll
  for (int j = 0; j < JSPLIT; ++j) {
    const float2 p = lsePart[(size_t)row * JSPLIT + j];
    lse_merge(M, S, p.x, p.y);
  }
  const int yv = y[row];
  const bf16* a  = zsbf + (size_t)row * E_;
  const bf16* vr = vocabbf + (size_t)yv * E_;
  float tgt = vocab_b[yv];
  for (int e = 0; e < E_; e += 8) {
    bf16x8 av = *(const bf16x8*)(a + e);
    bf16x8 vv = *(const bf16x8*)(vr + e);
#pragma unroll
    for (int i = 0; i < 8; ++i) tgt += (float)av[i] * (float)vv[i];
  }
  out[row] = tgt - (M + __logf(S));
}

// ---------------------------------------------------------------------------
// Host-side launcher
// ---------------------------------------------------------------------------
extern "C" void kernel_launch(void* const* d_in, const int* in_sizes, int n_in,
                              void* d_out, int out_size, void* d_ws, size_t ws_size,
                              hipStream_t stream) {
  const int*   zi     = (const int*)  d_in[0];
  const int*   y      = (const int*)  d_in[1];
  const float* noise  = (const float*)d_in[2];
  const float* latent = (const float*)d_in[3];
  const float* ekw = (const float*)d_in[4];  const float* ekb = (const float*)d_in[5];
  const float* evw = (const float*)d_in[6];  const float* evb = (const float*)d_in[7];
  const float* tkw = (const float*)d_in[8];  const float* tkb = (const float*)d_in[9];
  const float* tvw = (const float*)d_in[10]; const float* tvb = (const float*)d_in[11];
  const float* vw  = (const float*)d_in[12]; const float* vb  = (const float*)d_in[13];
  float* out = (float*)d_out;

  char* ws = (char*)d_ws;
  bf16*   z0bf  = (bf16*)ws;                                 // 64*512 bf16   (64KB)
  float*  zinit = (float*)(ws + 64 * 1024);                  // 64*512 f32    (128KB)
  bf16*   proj  = (bf16*)(ws + 192 * 1024);                  // 4*64*32768 bf16 (16MB)
  bf16*   zsbf  = proj + (size_t)4 * B_ * (M_ * E_);         // 4096*512 bf16 (4MB)
  bf16*   vwbf  = zsbf + (size_t)NROWS * E_;                 // 32000*512 bf16 (32MB)
  float2* lseP  = (float2*)(vwbf + (size_t)V_ * E_);         // 4096*8 float2 (256KB)

  prep_kernel<<<B_, 256, 0, stream>>>(zi, latent, noise, z0bf, zinit);

  const long nv = (long)V_ * E_;
  cvt_bf16_kernel<<<(unsigned)((nv / 4 + 255) / 256), 256, 0, stream>>>(vw, vwbf, nv);

  proj_kernel<<<dim3(64, 4), 256, 0, stream>>>(z0bf, ekw, evw, tkw, tvw,
                                               ekb, evb, tkb, tvb, proj);

  scan_kernel<<<B_, 256, 0, stream>>>(zinit, proj, zsbf);

  vocab_kernel<<<dim3(NROWS / 64, JSPLIT), 256, 0, stream>>>(zsbf, vwbf, vb, lseP);

  finish_kernel<<<(NROWS + 255) / 256, 256, 0, stream>>>(zsbf, vwbf, vb, y, lseP, out);
}
